// relationalGraphStack_936302871188
// MI455X (gfx1250) — compile-verified, matrix operand order checked by reading, not run on hardware
//
#include <hip/hip_runtime.h>
#include <hip/hip_bf16.h>

#define NN 50000
#define EE 800000
#define RRR 8
#define DDD 64

typedef __bf16 bf16x16 __attribute__((ext_vector_type(16)));
typedef __bf16 bf16x8  __attribute__((ext_vector_type(8)));
typedef float  f32x8   __attribute__((ext_vector_type(8)));

// native f32 -> bf16 (RNE); clang lowers to hardware cvt on gfx1250
__device__ __forceinline__ __bf16 f2bf(float f) { return (__bf16)f; }

__device__ __forceinline__ void fatomic_add(float* p, float v) {
    __hip_atomic_fetch_add(p, v, __ATOMIC_RELAXED, __HIP_MEMORY_SCOPE_AGENT);
}

// ---------------------------------------------------------------- zero fill
__global__ void zero_kernel(float* __restrict__ p, int n) {
    int i = blockIdx.x * blockDim.x + threadIdx.x;
    int stride = gridDim.x * blockDim.x;
    for (; i < n; i += stride) p[i] = 0.0f;
}

// ------------------------------------------------------- degree accumulation
__global__ void deg_kernel(const float* __restrict__ w, const int* __restrict__ nout,
                           const int* __restrict__ rel, float* __restrict__ deg) {
    int e = blockIdx.x * blockDim.x + threadIdx.x;
    if (e < EE) fatomic_add(&deg[nout[e] * RRR + rel[e]], w[e]);
}

__global__ void ew_kernel(const float* __restrict__ w, const int* __restrict__ nout,
                          const int* __restrict__ rel, const float* __restrict__ deg,
                          float* __restrict__ ew) {
    int e = blockIdx.x * blockDim.x + threadIdx.x;
    if (e < EE) ew[e] = w[e] / deg[nout[e] * RRR + rel[e]];
}

// ------------------------------------------- edge gather * ew -> segment sum
__global__ void scatter_kernel(const float* __restrict__ feat, const float* __restrict__ ew,
                               const int* __restrict__ nin, const int* __restrict__ nout,
                               const int* __restrict__ rel, float* __restrict__ upd) {
    int t = blockIdx.x * blockDim.x + threadIdx.x;
    if (t >= EE * DDD) return;
    int e = t >> 6, d = t & 63;
    int seg = nout[e] * RRR + rel[e];
    fatomic_add(&upd[(size_t)seg * DDD + d], feat[(size_t)nin[e] * DDD + d] * ew[e]);
}

// --------------------------------------------------------------- WMMA GEMM
// out[M,64] = [Aagg | Aself] (K = KA+64) x [Wl^T ; Ws^T]  + (bl + bs)
// one wave = one 16x64 output tile; 4 accumulators of v_wmma_f32_16x16x32_bf16
// B (<=74KB bf16) staged in LDS; per-feature BN sums fused into epilogue.
template <int KA, bool RELMODE>
__global__ __launch_bounds__(256) void gemm_wmma_kernel(
        const float* __restrict__ Aagg, const float* __restrict__ Aself,
        const float* __restrict__ Wl, const float* __restrict__ bl,
        const float* __restrict__ Ws, const float* __restrict__ bs,
        float* __restrict__ out, float* __restrict__ stats, int ntiles) {
    constexpr int KB = 64;
    constexpr int KT = KA + KB;
    constexpr int LDK = KT + 8;             // pad (mult of 8 -> 16B aligned rows)
    extern __shared__ __bf16 ldsB[];        // [64][LDK]

    // cooperative B fill:  B[k][n] = k<KA ? Wl[n][k] : Ws[n][k-KA]
    for (int idx = threadIdx.x; idx < 64 * KT; idx += blockDim.x) {
        int n = idx / KT, k = idx % KT;
        float v = (k < KA) ? Wl[n * KA + k] : Ws[n * KB + (k - KA)];
        ldsB[n * LDK + k] = f2bf(v);
    }
    __syncthreads();

    int wave = blockIdx.x * (blockDim.x >> 5) + (threadIdx.x >> 5);
    if (wave >= ntiles) return;
    int lane  = threadIdx.x & 31;
    int l16   = lane & 15;
    int khalf = lane >> 4;
    int m0    = wave * 16;
    int mrow  = m0 + l16;

    const float* pAgg;
    const float* pSelf;
    if constexpr (RELMODE) {                 // output row g = r*N + i (relation-major)
        int r = m0 / NN;                     // tiles never straddle r (N % 16 == 0)
        int i = mrow - r * NN;
        pAgg  = Aagg + ((size_t)i * RRR + r) * DDD;   // node-major segment buffer
        pSelf = Aself + (size_t)i * DDD;              // h tiled across relations
    } else {
        pAgg  = Aagg + (size_t)mrow * KA;    // [N, R*D] flat == [N*R, D] flat
        pSelf = Aself + (size_t)mrow * DDD;
    }

    f32x8 acc[4] = {};

    for (int kb = 0; kb < KT; kb += 32) {
        // A fragment: 16-bit A 16x32 layout -> lane holds K = {k0..k0+7, k1..k1+7}
        int k0 = kb + 8 * khalf;
        int k1 = k0 + 16;
        const float* p0 = (k0 < KA) ? (pAgg + k0) : (pSelf + (k0 - KA));
        const float* p1 = (k1 < KA) ? (pAgg + k1) : (pSelf + (k1 - KA));
        float4 A0 = *(const float4*)(p0);
        float4 A1 = *(const float4*)(p0 + 4);
        float4 A2 = *(const float4*)(p1);
        float4 A3 = *(const float4*)(p1 + 4);
        bf16x16 a;
        a[0]  = f2bf(A0.x); a[1]  = f2bf(A0.y); a[2]  = f2bf(A0.z); a[3]  = f2bf(A0.w);
        a[4]  = f2bf(A1.x); a[5]  = f2bf(A1.y); a[6]  = f2bf(A1.z); a[7]  = f2bf(A1.w);
        a[8]  = f2bf(A2.x); a[9]  = f2bf(A2.y); a[10] = f2bf(A2.z); a[11] = f2bf(A2.w);
        a[12] = f2bf(A3.x); a[13] = f2bf(A3.y); a[14] = f2bf(A3.z); a[15] = f2bf(A3.w);

        // issue all 8 LDS fragment loads before the 4 WMMAs so one dscnt wait
        // covers the whole group (lets ds latency overlap the matrix pipe)
        bf16x8 blo[4], bhi[4];
        #pragma unroll
        for (int nt = 0; nt < 4; ++nt) {
            const __bf16* bp = ldsB + (nt * 16 + l16) * LDK;
            blo[nt] = *(const bf16x8*)(bp + k0);
            bhi[nt] = *(const bf16x8*)(bp + k1);
        }
        #pragma unroll
        for (int nt = 0; nt < 4; ++nt) {
            bf16x16 b;
            #pragma unroll
            for (int j = 0; j < 8; ++j) { b[j] = blo[nt][j]; b[8 + j] = bhi[nt][j]; }
            acc[nt] = __builtin_amdgcn_wmma_f32_16x16x32_bf16(
                false, a, false, b, (short)0, acc[nt], false, false);
        }
    }

    // epilogue: bias, store, fused BN partial sums (sum / sumsq per feature)
    #pragma unroll
    for (int nt = 0; nt < 4; ++nt) {
        int n = nt * 16 + l16;
        float bias = bl[n] + bs[n];
        float s = 0.0f, s2 = 0.0f;
        #pragma unroll
        for (int v = 0; v < 8; ++v) {
            float val = acc[nt][v] + bias;
            size_t row = (size_t)(m0 + v + 8 * khalf);   // C layout: VGPR v -> M = v (+8)
            out[row * DDD + n] = val;
            s += val; s2 += val * val;
        }
        s  += __shfl_xor(s, 16, 32);     // combine the two lane-halves (same column)
        s2 += __shfl_xor(s2, 16, 32);
        if (khalf == 0) {
            fatomic_add(&stats[n], s);
            fatomic_add(&stats[DDD + n], s2);
        }
    }
}

// ---------------------------------------------------- BN (training) + ReLU
__global__ void bn_relu_kernel(float* __restrict__ data, const float* __restrict__ stats,
                               const float* __restrict__ gamma, const float* __restrict__ beta,
                               int total, float invM) {
    int i = blockIdx.x * blockDim.x + threadIdx.x;
    int stride = gridDim.x * blockDim.x;
    for (; i < total; i += stride) {
        int d = i & 63;
        float mean = stats[d] * invM;
        float var  = stats[64 + d] * invM - mean * mean;  // biased variance
        float inv  = rsqrtf(var + 1e-5f);
        float v = (data[i] - mean) * inv * gamma[d] + beta[d];
        data[i] = v > 0.0f ? v : 0.0f;
    }
}

extern "C" void kernel_launch(void* const* d_in, const int* in_sizes, int n_in,
                              void* d_out, int out_size, void* d_ws, size_t ws_size,
                              hipStream_t stream) {
    const float* x      = (const float*)d_in[0];
    const float* edge_w = (const float*)d_in[1];
    const float* Wlin1  = (const float*)d_in[2];
    const float* blin1  = (const float*)d_in[3];
    const float* Wself1 = (const float*)d_in[4];
    const float* bself1 = (const float*)d_in[5];
    const float* gamma1 = (const float*)d_in[6];
    const float* beta1  = (const float*)d_in[7];
    const float* Wlin2  = (const float*)d_in[8];
    const float* blin2  = (const float*)d_in[9];
    const float* Wself2 = (const float*)d_in[10];
    const float* bself2 = (const float*)d_in[11];
    const float* gamma2 = (const float*)d_in[12];
    const float* beta2  = (const float*)d_in[13];
    const int*   nin    = (const int*)d_in[14];
    const int*   nout   = (const int*)d_in[15];
    const int*   rel    = (const int*)d_in[16];
    (void)in_sizes; (void)n_in; (void)out_size; (void)ws_size;

    float* ws    = (float*)d_ws;
    float* deg   = ws;                       // 400,000
    float* ew    = ws + 400000;              // 800,000
    float* upd   = ws + 1200000;             // 25,600,000 (reused both layers)
    float* h     = ws + 26800000;            // 3,200,000
    float* stats = ws + 30000000;            // 256 (sum1,sq1,sum2,sq2)
    float* outp  = (float*)d_out;            // [R, N, D] = 25,600,000

    // init accumulators (harness does not re-zero between replays)
    zero_kernel<<<512, 256, 0, stream>>>(deg, NN * RRR);
    zero_kernel<<<1, 256, 0, stream>>>(stats, 256);
    zero_kernel<<<2048, 256, 0, stream>>>(upd, NN * RRR * DDD);

    int eb = (EE + 255) / 256;
    deg_kernel<<<eb, 256, 0, stream>>>(edge_w, nout, rel, deg);
    ew_kernel<<<eb, 256, 0, stream>>>(edge_w, nout, rel, deg, ew);

    int sb = (EE * DDD + 255) / 256;
    // ---- layer 1 ----
    scatter_kernel<<<sb, 256, 0, stream>>>(x, ew, nin, nout, rel, upd);
    {
        int ntiles = NN / 16;                          // 3125
        int blocks = (ntiles + 7) / 8;                 // 8 waves / block
        size_t sh = (size_t)64 * (512 + 64 + 8) * sizeof(__bf16);   // 74,752 B
        gemm_wmma_kernel<512, false><<<blocks, 256, sh, stream>>>(
            upd, x, Wlin1, blin1, Wself1, bself1, h, stats, ntiles);
    }
    bn_relu_kernel<<<2048, 256, 0, stream>>>(h, stats, gamma1, beta1,
                                             NN * DDD, 1.0f / (float)NN);

    // ---- layer 2 ----
    zero_kernel<<<2048, 256, 0, stream>>>(upd, NN * RRR * DDD);
    scatter_kernel<<<sb, 256, 0, stream>>>(h, ew, nin, nout, rel, upd);
    {
        int ntiles = (NN * RRR) / 16;                  // 25,000
        int blocks = (ntiles + 7) / 8;
        size_t sh = (size_t)64 * (64 + 64 + 8) * sizeof(__bf16);    // 17,408 B
        gemm_wmma_kernel<64, true><<<blocks, 256, sh, stream>>>(
            upd, h, Wlin2, blin2, Wself2, bself2, outp, stats + 128, ntiles);
    }
    bn_relu_kernel<<<4096, 256, 0, stream>>>(outp, stats + 128, gamma2, beta2,
                                             NN * RRR * DDD, 1.0f / (float)(NN * RRR));
}
